// KANBlock_23321672417777
// MI455X (gfx1250) — matrix operand order repaired and checked by reading, not compiled, not run on hardware
//
#include <hip/hip_runtime.h>

// ---------------- KAN constants (from reference) ----------------
#define GSPLINE 5
#define KORD    3
#define NKNOT   (GSPLINE + 2*KORD + 1)   // 12 knots
#define NB0     (GSPLINE + 2*KORD)       // 11 order-0 bases
#define GK      (GSPLINE + KORD)         // 8 final bases
#define FEAT    (GK + 1)                 // 1 silu + 8 bases = 9 features/channel

static constexpr int D_MODEL = 512;
static constexpr int D_FF    = 2048;
static constexpr int NTOK    = 2 * 2048; // B*S

typedef __bf16 bf16;
typedef __attribute__((ext_vector_type(16))) __bf16 v16bf;
typedef __attribute__((ext_vector_type(8)))  __bf16 v8bf;
typedef __attribute__((ext_vector_type(2)))  __bf16 v2bf;
typedef __attribute__((ext_vector_type(8)))  float  v8f;

// Feature-major K layout: K = f*in + i  (f=0: silu base, f=1..8: spline bases).
// A[n][f*in+i] and W[o][f*in+i] use the same permutation, so the GEMM is unchanged.

// ---------------- weight packing: fp32 (out,in)+(out,in,8) -> bf16 (out, 9*in)
// One thread per channel PAIR -> every store is a coalesced 4B (2 x bf16).
__global__ void pack_weights(const float* __restrict__ wbase,
                             const float* __restrict__ wspl,
                             bf16* __restrict__ W, int in, long totpairs) {
  long tid = blockIdx.x * (long)blockDim.x + threadIdx.x;
  if (tid >= totpairs) return;
  const int  halfin = in >> 1;
  const long o  = tid / halfin;
  const int  i  = (int)(tid % halfin) * 2;
  const long e0 = o * in + i;                 // element (o,i)
  bf16* dst = W + o * (long)(FEAT * in) + i;  // row o, feature-major
  {
    v2bf p; p[0] = (bf16)wbase[e0]; p[1] = (bf16)wbase[e0 + 1];
    *(v2bf*)dst = p;
  }
  const float* s0 = wspl + e0 * GK;           // 8 floats for channel i
  const float* s1 = s0 + GK;                  // 8 floats for channel i+1
#pragma unroll
  for (int g = 0; g < GK; ++g) {
    v2bf p; p[0] = (bf16)s0[g]; p[1] = (bf16)s1[g];
    *(v2bf*)(dst + (long)(g + 1) * in) = p;
  }
}

// ---------------- featurize: x (ntok,in) fp32 -> A (ntok, 9*in) bf16
// One thread per channel pair; 9 coalesced 4B stores per thread.
__device__ __forceinline__ void kan_feats(float x, const float* g, float* out9) {
  out9[0] = x / (1.0f + __expf(-x));          // silu residual path
  float b[NB0];
#pragma unroll
  for (int t = 0; t < NB0; ++t)
    b[t] = (x >= g[t] && x < g[t + 1]) ? 1.0f : 0.0f;
#pragma unroll
  for (int j = 1; j <= KORD; ++j) {
#pragma unroll
    for (int t = 0; t < NB0 - j; ++t) {
      float left  = (x - g[t]) / (g[t + j] - g[t]);
      float right = (g[t + j + 1] - x) / (g[t + j + 1] - g[t + 1]);
      b[t] = left * b[t] + right * b[t + 1];
    }
  }
#pragma unroll
  for (int q = 0; q < GK; ++q) out9[1 + q] = b[q];
}

__global__ void featurize(const float* __restrict__ X,
                          const float* __restrict__ grid,
                          bf16* __restrict__ A, int in, long totpairs) {
  long tid = blockIdx.x * (long)blockDim.x + threadIdx.x;
  if (tid >= totpairs) return;
  float g[NKNOT];
#pragma unroll
  for (int m = 0; m < NKNOT; ++m) g[m] = grid[m];

  const int  halfin = in >> 1;
  const long n = tid / halfin;
  const int  i = (int)(tid % halfin) * 2;

  float f0[FEAT], f1[FEAT];
  kan_feats(X[n * in + i], g, f0);
  kan_feats(X[n * in + i + 1], g, f1);

  bf16* dst = A + n * (long)(FEAT * in) + i;
#pragma unroll
  for (int f = 0; f < FEAT; ++f) {
    v2bf p; p[0] = (bf16)f0[f]; p[1] = (bf16)f1[f];
    *(v2bf*)(dst + (long)f * in) = p;
  }
}

// ---------------- bf16 WMMA GEMM: C[ntok][nout] = A[ntok][K] * W[nout][K]^T
__device__ __forceinline__ void load_frags(const bf16* Arow0, const bf16* Arow1,
                                           const bf16* const (&Wcol)[4],
                                           int k0, int aoff, int boff,
                                           v16bf (&a)[2], v16bf (&b)[4]) {
  v8bf lo, hi;
  lo = *(const v8bf*)(Arow0 + k0 + aoff);
  hi = *(const v8bf*)(Arow0 + k0 + 16 + aoff);
#pragma unroll
  for (int t = 0; t < 8; ++t) { a[0][t] = lo[t]; a[0][8 + t] = hi[t]; }
  lo = *(const v8bf*)(Arow1 + k0 + aoff);
  hi = *(const v8bf*)(Arow1 + k0 + 16 + aoff);
#pragma unroll
  for (int t = 0; t < 8; ++t) { a[1][t] = lo[t]; a[1][8 + t] = hi[t]; }
#pragma unroll
  for (int n = 0; n < 4; ++n) {
    lo = *(const v8bf*)(Wcol[n] + k0 + boff);
    hi = *(const v8bf*)(Wcol[n] + k0 + boff + 16);
#pragma unroll
    for (int t = 0; t < 8; ++t) { b[n][t] = lo[t]; b[n][8 + t] = hi[t]; }
  }
}

__device__ __forceinline__ void mma8(const v16bf (&a)[2], const v16bf (&b)[4],
                                     v8f (&acc)[2][4]) {
#pragma unroll
  for (int m = 0; m < 2; ++m)
#pragma unroll
    for (int n = 0; n < 4; ++n)
      acc[m][n] = __builtin_amdgcn_wmma_f32_16x16x32_bf16(
          false, a[m], false, b[n], (short)0, acc[m][n], false, false);
}

// block = 256 thr (8 waves), block tile 128x128; wave tile 32x64 = 2x4 WMMA tiles.
// K loop double-buffered in registers (K1=4608, K2=18432; both multiples of 64).
__global__ void __launch_bounds__(256)
kan_gemm(const bf16* __restrict__ A, const bf16* __restrict__ W,
         float* __restrict__ C, int nout, int kdim) {
  const int lane = threadIdx.x & 31;
  const int wave = threadIdx.x >> 5;
  const int half = lane >> 4;
  const int l16  = lane & 15;

  const int wm = wave & 3;
  const int wn = wave >> 2;

  const long row0 = (long)blockIdx.y * 128 + wm * 32;   // token base (2 x 16 rows)
  const int  col0 = blockIdx.x * 128 + wn * 64;         // out base  (4 x 16 cols)

  const bf16* Arow0 = A + (row0 + l16) * (long)kdim;
  const bf16* Arow1 = A + (row0 + 16 + l16) * (long)kdim;
  const bf16* Wcol[4];
#pragma unroll
  for (int n = 0; n < 4; ++n)
    Wcol[n] = W + (long)(col0 + n * 16 + l16) * kdim;

  v8f acc[2][4];
#pragma unroll
  for (int m = 0; m < 2; ++m)
#pragma unroll
    for (int n = 0; n < 4; ++n)
#pragma unroll
      for (int r = 0; r < 8; ++r) acc[m][n][r] = 0.0f;

  const int aoff = half * 8;    // A frag: K = aoff+0..7 and 16+aoff+0..7
  const int boff = half * 8;    // B frag: K = boff+0..7 and 16+boff+0..7

  v16bf a0[2], b0[4], a1[2], b1[4];
  load_frags(Arow0, Arow1, Wcol, 0, aoff, boff, a0, b0);
  load_frags(Arow0, Arow1, Wcol, 32, aoff, boff, a1, b1);

  int k0 = 0;
  for (; k0 + 64 < kdim; k0 += 64) {
    mma8(a0, b0, acc);
    load_frags(Arow0, Arow1, Wcol, k0 + 64, aoff, boff, a0, b0);
    mma8(a1, b1, acc);
    load_frags(Arow0, Arow1, Wcol, k0 + 96, aoff, boff, a1, b1);
  }
  mma8(a0, b0, acc);
  mma8(a1, b1, acc);

  // epilogue: lane l16 = output col, reg r = row (+8 for lanes 16-31)
#pragma unroll
  for (int m = 0; m < 2; ++m) {
#pragma unroll
    for (int n = 0; n < 4; ++n) {
      float* cp = C + (row0 + m * 16 + half * 8) * (long)nout + col0 + n * 16 + l16;
#pragma unroll
      for (int r = 0; r < 8; ++r)
        cp[(long)r * nout] = acc[m][n][r];
    }
  }
}

// ---------------- host-side orchestration ----------------
extern "C" void kernel_launch(void* const* d_in, const int* in_sizes, int n_in,
                              void* d_out, int out_size, void* d_ws, size_t ws_size,
                              hipStream_t stream) {
  const float* x       = (const float*)d_in[0];
  const float* grid    = (const float*)d_in[1];
  const float* w1_base = (const float*)d_in[2];
  const float* w1_spl  = (const float*)d_in[3];
  const float* w2_base = (const float*)d_in[4];
  const float* w2_spl  = (const float*)d_in[5];

  const long K1 = (long)D_MODEL * FEAT;  // 4608
  const long K2 = (long)D_FF * FEAT;     // 18432

  char* ws = (char*)d_ws;
  size_t off = 0;
  auto alloc = [&](size_t bytes) {
    void* p = ws + off;
    off = (off + bytes + 255) & ~(size_t)255;
    return p;
  };
  bf16*  A1 = (bf16*)alloc((size_t)NTOK * K1 * sizeof(bf16));
  bf16*  W1 = (bf16*)alloc((size_t)D_FF * K1 * sizeof(bf16));
  float* H  = (float*)alloc((size_t)NTOK * D_FF * sizeof(float));
  bf16*  A2 = (bf16*)alloc((size_t)NTOK * K2 * sizeof(bf16));
  bf16*  W2 = (bf16*)alloc((size_t)D_MODEL * K2 * sizeof(bf16));

  // 1-2) pack weights to bf16 in fused feature-major (out, 9*in) layout
  {
    long tp = (long)D_FF * D_MODEL / 2;
    pack_weights<<<(tp + 255) / 256, 256, 0, stream>>>(w1_base, w1_spl, W1, D_MODEL, tp);
    pack_weights<<<(tp + 255) / 256, 256, 0, stream>>>(w2_base, w2_spl, W2, D_FF, tp);
  }
  // 3) featurize layer-1 input
  {
    long tp = (long)NTOK * D_MODEL / 2;
    featurize<<<(tp + 255) / 256, 256, 0, stream>>>(x, grid, A1, D_MODEL, tp);
  }
  // 4) layer-1 GEMM: (4096 x 4608) x (2048 x 4608)^T -> H
  {
    dim3 g(D_FF / 128, NTOK / 128);
    kan_gemm<<<g, 256, 0, stream>>>(A1, W1, H, D_FF, (int)K1);
  }
  // 5) featurize layer-2 input
  {
    long tp = (long)NTOK * D_FF / 2;
    featurize<<<(tp + 255) / 256, 256, 0, stream>>>(H, grid, A2, D_FF, tp);
  }
  // 6) layer-2 GEMM: (4096 x 18432) x (512 x 18432)^T -> out
  {
    dim3 g(D_MODEL / 128, NTOK / 128);
    kan_gemm<<<g, 256, 0, stream>>>(A2, W2, (float*)d_out, D_MODEL, (int)K2);
  }
}
